// LSTMTagger_65481071405716
// MI455X (gfx1250) — matrix operand order repaired and checked by reading, not compiled, not run on hardware
//
#include <hip/hip_runtime.h>
#include <hip/hip_bf16.h>

// ---------------------------------------------------------------------------
// Problem constants (from reference)
// ---------------------------------------------------------------------------
#define SEQ      4096
#define EMBED    4
#define HIDDEN   1024
#define GATES    (4 * HIDDEN)          // 4096
#define LSTM_IN  6
#define VOCAB    50257
#define NTAG     50257
#define NPAD     50688                 // multiple of 128, >= NTAG

// Workspace layout (bytes, all 256-aligned)
static constexpr size_t O_CTRL  = 0;                               // cnt, gen, mean
static constexpr size_t O_FEATS = 256;                             // 4096*6 f32
static constexpr size_t O_HS    = O_FEATS + (size_t)SEQ * LSTM_IN * 4;      // 98560
static constexpr size_t O_HSB   = O_HS    + (size_t)SEQ * HIDDEN * 4;       // +16 MB
static constexpr size_t O_WB    = O_HSB   + (size_t)SEQ * HIDDEN * 2;       // +8 MB
// O_WB + NPAD*HIDDEN*2 ~= 129 MB total

// ---------------------------------------------------------------------------
// Helpers
// ---------------------------------------------------------------------------
typedef __attribute__((ext_vector_type(16))) __bf16 v16bf;
typedef __attribute__((ext_vector_type(8)))  float  v8f;

union FragBF { v16bf v; uint4 q[2]; };

__device__ __forceinline__ float sigmoidf(float x) { return 1.0f / (1.0f + __expf(-x)); }

__device__ __forceinline__ __bf16 f32_to_bf16(float f) {
  unsigned u = __builtin_bit_cast(unsigned, f);
  unsigned short r = (unsigned short)((u + 0x7FFFu + ((u >> 16) & 1u)) >> 16);
  return __builtin_bit_cast(__bf16, r);
}

// Grid-wide barrier for the persistent LSTM kernel.
__device__ __forceinline__ void grid_barrier(int* cnt, int* gen) {
  __syncthreads();
  if (threadIdx.x == 0) {
    __threadfence();
    int g = __hip_atomic_load(gen, __ATOMIC_ACQUIRE, __HIP_MEMORY_SCOPE_AGENT);
    int prev = __hip_atomic_fetch_add(cnt, 1, __ATOMIC_ACQ_REL, __HIP_MEMORY_SCOPE_AGENT);
    if (prev == (int)gridDim.x - 1) {
      __hip_atomic_store(cnt, 0, __ATOMIC_RELAXED, __HIP_MEMORY_SCOPE_AGENT);
      __hip_atomic_fetch_add(gen, 1, __ATOMIC_ACQ_REL, __HIP_MEMORY_SCOPE_AGENT);
    } else {
      while (__hip_atomic_load(gen, __ATOMIC_ACQUIRE, __HIP_MEMORY_SCOPE_AGENT) == g) {
        __builtin_amdgcn_s_sleep(8);
      }
    }
  }
  __syncthreads();
}

// ---------------------------------------------------------------------------
// K0: reset control words each launch (determinism across graph replays)
// ---------------------------------------------------------------------------
__global__ void k_init(int* ctrl) {
  ctrl[0] = 0;
  ctrl[1] = 0;
  ((float*)ctrl)[2] = 0.0f;
}

// ---------------------------------------------------------------------------
// K1: embedding gather + conv sigmoid  ->  feats[:, 0:5]
// ---------------------------------------------------------------------------
__global__ void k_embed_conv(const int* __restrict__ sent,
                             const float* __restrict__ emb,
                             const float* __restrict__ conv_w,
                             const float* __restrict__ conv_b,
                             float* __restrict__ feats) {
  int s = blockIdx.x * blockDim.x + threadIdx.x;
  if (s >= SEQ) return;
  int tok = sent[s];
  float e0 = emb[tok * EMBED + 0], e1 = emb[tok * EMBED + 1];
  float e2 = emb[tok * EMBED + 2], e3 = emb[tok * EMBED + 3];
  feats[s * LSTM_IN + 0] = e0;
  feats[s * LSTM_IN + 1] = e1;
  feats[s * LSTM_IN + 2] = e2;
  feats[s * LSTM_IN + 3] = e3;
  float z = e0 * conv_w[0] + e1 * conv_w[1] + e2 * conv_w[2] + e3 * conv_w[3] + conv_b[0];
  feats[s * LSTM_IN + 4] = sigmoidf(z);
}

// K2: deterministic mean of conv feature (single block, fixed-order tree)
__global__ void k_mean(const float* __restrict__ feats, float* __restrict__ meanOut) {
  __shared__ float red[256];
  float acc = 0.0f;
  for (int s = threadIdx.x; s < SEQ; s += 256) acc += feats[s * LSTM_IN + 4];
  red[threadIdx.x] = acc;
  __syncthreads();
  for (int off = 128; off > 0; off >>= 1) {
    if (threadIdx.x < off) red[threadIdx.x] += red[threadIdx.x + off];
    __syncthreads();
  }
  if (threadIdx.x == 0) meanOut[0] = red[0] * (1.0f / (float)SEQ);
}

// K3: broadcast attn column
__global__ void k_attn(float* __restrict__ feats, const float* __restrict__ meanIn) {
  int s = blockIdx.x * blockDim.x + threadIdx.x;
  if (s < SEQ) feats[s * LSTM_IN + 5] = meanIn[0];
}

// ---------------------------------------------------------------------------
// K4: persistent LSTM scan. 64 blocks x 256 threads; c-state in LDS;
//     4 lanes per gate-row, shuffle reduce, grid barrier per step.
// ---------------------------------------------------------------------------
__global__ void k_lstm(const float* __restrict__ feats,
                       const float* __restrict__ w_ih,
                       const float* __restrict__ w_hh,
                       const float* __restrict__ b_lstm,
                       float* __restrict__ hs,
                       int* ctrl) {
  __shared__ float z_s[64];
  __shared__ float c_s[16];
  __shared__ float x_s[LSTM_IN];

  const int tid       = threadIdx.x;
  const int b         = blockIdx.x;
  const int row_local = tid >> 2;            // 0..63
  const int part      = tid & 3;             // 0..3
  const int gate      = row_local >> 4;      // i, f, g, o
  const int j_local   = row_local & 15;
  const int row       = gate * HIDDEN + b * 16 + j_local;   // 0..4095

  const float* wrow = w_hh + (size_t)row * HIDDEN + part * 256;
  if (tid < 16) c_s[tid] = 0.0f;

  int* cnt = ctrl + 0;
  int* gen = ctrl + 1;

  for (int t = 0; t < SEQ; ++t) {
    if (tid < LSTM_IN) x_s[tid] = feats[t * LSTM_IN + tid];
    __syncthreads();

    float acc = 0.0f;
    if (t > 0) {
      const float* hprev = hs + (size_t)(t - 1) * HIDDEN + part * 256;
      #pragma unroll 8
      for (int k = 0; k < 256; ++k) acc += wrow[k] * hprev[k];
    }
    acc += __shfl_xor(acc, 1, 32);
    acc += __shfl_xor(acc, 2, 32);

    if (part == 0) {
      const float* wi = w_ih + (size_t)row * LSTM_IN;
      float zx = b_lstm[row];
      #pragma unroll
      for (int c = 0; c < LSTM_IN; ++c) zx += wi[c] * x_s[c];
      z_s[row_local] = acc + zx;
    }
    __syncthreads();

    if (tid < 16) {
      int j = tid;
      float zi = z_s[j], zf = z_s[16 + j], zg = z_s[32 + j], zo = z_s[48 + j];
      float c = sigmoidf(zf) * c_s[j] + sigmoidf(zi) * tanhf(zg);
      c_s[j] = c;
      hs[(size_t)t * HIDDEN + b * 16 + j] = sigmoidf(zo) * tanhf(c);
    }
    grid_barrier(cnt, gen);
  }
}

// ---------------------------------------------------------------------------
// K5/K6: f32 -> bf16 conversions (w_out zero-padded to NPAD rows)
// ---------------------------------------------------------------------------
__global__ void k_cvt_hs(const float* __restrict__ hs, __bf16* __restrict__ hsb) {
  int i = blockIdx.x * blockDim.x + threadIdx.x;
  if (i < SEQ * HIDDEN) hsb[i] = f32_to_bf16(hs[i]);
}

__global__ void k_cvt_w(const float* __restrict__ w_out, __bf16* __restrict__ wb) {
  long long i = (long long)blockIdx.x * blockDim.x + threadIdx.x;
  if (i >= (long long)NPAD * HIDDEN) return;
  int n = (int)(i >> 10);
  int k = (int)(i & 1023);
  float v = (n < NTAG) ? w_out[(size_t)n * HIDDEN + k] : 0.0f;
  wb[i] = f32_to_bf16(v);
}

// ---------------------------------------------------------------------------
// K7: logits = hs_bf16 @ w_out_bf16^T + b_out via v_wmma_f32_16x16x32_bf16.
//   Block tile 128(M) x 128(N); 8 waves, wave w owns M-tile mBase + w*16 and
//   all 8 N-subtiles (8 accumulators). B tile (128 rows x 32 k bf16 = 8 KB)
//   is staged once per block per k-step into LDS with
//   GLOBAL_LOAD_ASYNC_TO_LDS_B128 (double-buffered, s_wait_asynccnt), giving
//   ~64 flop/byte from L2 instead of 13.
//   LDS row stride 80 B (16B-aligned; 20-dword stride -> conflict-free
//   ds_load_b128 fragment reads).
//   Fragment layouts used here (A/B k order is this code's own, D per ISA):
//     A 16x32 bf16: lane m=lane&15; elems 0-15 K=half*16+j (contiguous)
//     B 32x16 bf16: lane n=lane&15; elems 0-15 K=half*16+j (contiguous)
//     D f32:        VGPR r -> row m=r+8*half, col n=lane&15
// ---------------------------------------------------------------------------
#define BROW 40                                  // LDS row stride in bf16 (80 B)

__global__ void k_gemm_wmma(const __bf16* __restrict__ hsb,
                            const __bf16* __restrict__ wb,
                            const float* __restrict__ b_out,
                            float* __restrict__ out) {
  __shared__ __align__(16) __bf16 btile[2][128 * BROW];

  const int tid  = threadIdx.x;
  const int lane = tid & 31;
  const int wave = tid >> 5;                 // 0..7
  const int lh   = (lane >> 4) & 1;
  const int nl   = lane & 15;

  const int mTile = blockIdx.y * 128 + wave * 16;
  const int nBase = blockIdx.x * 128;

  // --- async staging coordinates: thread i moves 32 B of the 8 KB B tile ---
  const int srow  = tid >> 1;                // 0..127 : row within tile
  const int shalf = tid & 1;                 // which 32B half of the 64B row
  const unsigned ldsRowOff = (unsigned)(srow * (BROW * 2) + shalf * 32);
  const __bf16* gsrc = wb + (size_t)(nBase + srow) * HIDDEN + shalf * 16;

  v8f acc[8] = {};
  const __bf16* aRow = hsb + (size_t)(mTile + nl) * HIDDEN;

  // stage k-block 0 into buffer 0
  {
    unsigned lds = (unsigned)(size_t)(&btile[0][0]) + ldsRowOff;
    unsigned long long ga = (unsigned long long)(size_t)gsrc;
    asm volatile(
        "global_load_async_to_lds_b128 %0, %1, off\n\t"
        "global_load_async_to_lds_b128 %0, %1, off offset:16"
        :: "v"(lds), "v"(ga) : "memory");
  }

  for (int k0 = 0; k0 < HIDDEN; k0 += 32) {
    const int buf  = (k0 >> 5) & 1;
    const bool more = (k0 + 32) < HIDDEN;

    if (more) {  // prefetch next k-block into the other buffer
      unsigned lds = (unsigned)(size_t)(&btile[buf ^ 1][0]) + ldsRowOff;
      unsigned long long ga = (unsigned long long)(size_t)(gsrc + (k0 + 32));
      asm volatile(
          "global_load_async_to_lds_b128 %0, %1, off\n\t"
          "global_load_async_to_lds_b128 %0, %1, off offset:16"
          :: "v"(lds), "v"(ga) : "memory");
      asm volatile("s_wait_asynccnt 0x2" ::: "memory");   // current buf done
    } else {
      asm volatile("s_wait_asynccnt 0x0" ::: "memory");
    }
    __syncthreads();   // all waves' portions of buf are in LDS

    FragBF a;
    a.q[0] = *(const uint4*)(aRow + k0 + lh * 16);
    a.q[1] = *(const uint4*)(aRow + k0 + lh * 16 + 8);

    const __bf16* bb = &btile[buf][0] + (size_t)nl * BROW + lh * 16;
    #pragma unroll
    for (int t = 0; t < 8; ++t) {
      FragBF bf;
      const uint4* bp = (const uint4*)(bb + t * 16 * BROW);
      bf.q[0] = bp[0];
      bf.q[1] = bp[1];
      acc[t] = __builtin_amdgcn_wmma_f32_16x16x32_bf16(
          false, a.v, false, bf.v, (short)0, acc[t], false, false);
    }
    __syncthreads();   // done reading buf before it is restaged
  }

  #pragma unroll
  for (int t = 0; t < 8; ++t) {
    int n = nBase + t * 16 + nl;
    if (n < NTAG) {
      float bias = b_out[n];
      #pragma unroll
      for (int r = 0; r < 8; ++r) {
        int m = mTile + r + 8 * lh;
        out[(size_t)m * NTAG + n] = acc[t][r] + bias;
      }
    }
  }
}

// ---------------------------------------------------------------------------
// K8: in-place log_softmax per row (deterministic LDS tree reductions)
// ---------------------------------------------------------------------------
__global__ void k_logsoftmax(float* __restrict__ out) {
  __shared__ float red[256];
  const int m = blockIdx.x;
  float* row = out + (size_t)m * NTAG;

  float mx = -3.4e38f;
  for (int i = threadIdx.x; i < NTAG; i += 256) mx = fmaxf(mx, row[i]);
  red[threadIdx.x] = mx;
  __syncthreads();
  for (int off = 128; off > 0; off >>= 1) {
    if (threadIdx.x < off) red[threadIdx.x] = fmaxf(red[threadIdx.x], red[threadIdx.x + off]);
    __syncthreads();
  }
  const float rmax = red[0];
  __syncthreads();

  float s = 0.0f;
  for (int i = threadIdx.x; i < NTAG; i += 256) s += __expf(row[i] - rmax);
  red[threadIdx.x] = s;
  __syncthreads();
  for (int off = 128; off > 0; off >>= 1) {
    if (threadIdx.x < off) red[threadIdx.x] += red[threadIdx.x + off];
    __syncthreads();
  }
  const float lse = rmax + __logf(red[0]);
  __syncthreads();

  for (int i = threadIdx.x; i < NTAG; i += 256) row[i] = row[i] - lse;
}

// ---------------------------------------------------------------------------
// Host launcher
// ---------------------------------------------------------------------------
extern "C" void kernel_launch(void* const* d_in, const int* in_sizes, int n_in,
                              void* d_out, int out_size, void* d_ws, size_t ws_size,
                              hipStream_t stream) {
  (void)in_sizes; (void)n_in; (void)out_size; (void)ws_size;

  const int*   sent    = (const int*)d_in[0];
  const float* emb     = (const float*)d_in[1];
  const float* conv_w  = (const float*)d_in[2];
  const float* conv_b  = (const float*)d_in[3];
  const float* w_ih    = (const float*)d_in[4];
  const float* w_hh    = (const float*)d_in[5];
  const float* b_lstm  = (const float*)d_in[6];
  const float* w_out   = (const float*)d_in[7];
  const float* b_out   = (const float*)d_in[8];
  float*       out     = (float*)d_out;

  char* ws = (char*)d_ws;
  int*    ctrl  = (int*)(ws + O_CTRL);
  float*  mean  = (float*)(ws + O_CTRL) + 2;
  float*  feats = (float*)(ws + O_FEATS);
  float*  hs    = (float*)(ws + O_HS);
  __bf16* hsb   = (__bf16*)(ws + O_HSB);
  __bf16* wb    = (__bf16*)(ws + O_WB);

  k_init<<<1, 1, 0, stream>>>(ctrl);
  k_embed_conv<<<SEQ / 256, 256, 0, stream>>>(sent, emb, conv_w, conv_b, feats);
  k_mean<<<1, 256, 0, stream>>>(feats, mean);
  k_attn<<<SEQ / 256, 256, 0, stream>>>(feats, mean);

  k_lstm<<<64, 256, 0, stream>>>(feats, w_ih, w_hh, b_lstm, hs, ctrl);

  k_cvt_hs<<<(SEQ * HIDDEN) / 256, 256, 0, stream>>>(hs, hsb);
  k_cvt_w<<<((size_t)NPAD * HIDDEN) / 256, 256, 0, stream>>>(w_out, wb);

  dim3 gg(NPAD / 128, SEQ / 128);
  k_gemm_wmma<<<gg, 256, 0, stream>>>(hsb, wb, b_out, out);

  k_logsoftmax<<<SEQ, 256, 0, stream>>>(out);
}
